// MemorizingGPT_63702954934817
// MI455X (gfx1250) — matrix operand order repaired and checked by reading, not compiled, not run on hardware
//
#include <hip/hip_runtime.h>

typedef __bf16 bf16;
typedef __attribute__((ext_vector_type(16))) __bf16 v16bf;
typedef __attribute__((ext_vector_type(8)))  __bf16 v8bf;
typedef __attribute__((ext_vector_type(8)))  float  v8f;

union V16U { v16bf v; v8bf h[2]; };

constexpr int Bc = 2, Tc = 2048, Cc = 1024, Hc = 16, Dc = 64;
constexpr int Mrows = Bc * Tc; // 4096

// async global->LDS copy of 16 bytes per lane (CDNA5, ASYNCcnt-tracked)
#define ASYNC_CP16(ldsoff, gaddr)                                              \
  asm volatile("global_load_async_to_lds_b128 %0, %1, off"                     \
               :: "v"(ldsoff), "v"(gaddr) : "memory")
#define WAIT_ASYNC0() asm volatile("s_wait_asynccnt 0x0" ::: "memory")
#define WAIT_ASYNC4() asm volatile("s_wait_asynccnt 0x4" ::: "memory")

static __device__ __forceinline__ v8f wmma_bf16(v16bf a, v16bf b, v8f c) {
  return __builtin_amdgcn_wmma_f32_16x16x32_bf16(false, a, false, b, (short)0, c, false, false);
}

// ---------------- conversion kernels ----------------
__global__ void k_f32_to_bf16(const float* __restrict__ s, bf16* __restrict__ d, int n) {
  int i = blockIdx.x * blockDim.x + threadIdx.x;
  if (i < n) d[i] = (bf16)s[i];
}

// dst[n][k] = (bf16) src[k][n];  src is R x Ccol, dst is Ccol x R (k-contiguous)
__global__ __launch_bounds__(256)
void k_transpose_bf16(const float* __restrict__ src, bf16* __restrict__ dst, int R, int Ccol) {
  __shared__ float tile[32][33];
  int x = blockIdx.x * 32 + threadIdx.x;
  int y0 = blockIdx.y * 32;
  for (int j = threadIdx.y; j < 32; j += 8)
    tile[j][threadIdx.x] = src[(size_t)(y0 + j) * Ccol + x];
  __syncthreads();
  for (int j = threadIdx.y; j < 32; j += 8)
    dst[(size_t)(blockIdx.x * 32 + j) * R + (y0 + threadIdx.x)] = (bf16)tile[threadIdx.x][j];
}

// ---------------- tiled WMMA GEMM (double-buffered async-LDS staging) ----------------
// A: [M][K] bf16 (k contig). Bt: [N][K] bf16 (k contig).
// MODE 0: qkv epilogue: q (pre-scaled by 0.125) and k head-major [bh][t][d];
//         v transposed [bh][d][t].
// MODE 1: fp32 out + bias
template<int MODE>
__global__ __launch_bounds__(256)
void k_wmma_gemm(const bf16* __restrict__ A, const bf16* __restrict__ Bt,
                 const float* __restrict__ bias, int Nsz, int Ksz,
                 float* __restrict__ outf,
                 bf16* __restrict__ qout, bf16* __restrict__ kout, bf16* __restrict__ vout)
{
  __shared__ bf16 As[2][128][40];
  __shared__ bf16 Bs[2][128][40];
  int nblk = Nsz >> 7;
  int mb = blockIdx.x / nblk;
  int nb = blockIdx.x % nblk;
  int m0 = mb << 7, n0 = nb << 7;
  int tid = threadIdx.x;
  int w = tid >> 5, lane = tid & 31;
  int wm = w & 3, wn = w >> 2;          // 4 (M) x 2 (N) waves
  bool loh = lane < 16;
  int q16 = lane & 15;
  int abase = loh ? 0 : 8;
  int kboff = loh ? 0 : 8;

  v8f acc[2][4];
#pragma unroll
  for (int r = 0; r < 2; ++r)
#pragma unroll
    for (int c = 0; c < 4; ++c)
#pragma unroll
      for (int e = 0; e < 8; ++e) acc[r][c][e] = 0.f;

  int lr = tid >> 1;
  int lseg = (tid & 1) << 4;
  const bf16* Arow = A + (size_t)(m0 + lr) * Ksz + lseg;
  const bf16* Brow = Bt + (size_t)(n0 + lr) * Ksz + lseg;
  unsigned ldsA[2], ldsB[2];
  ldsA[0] = (unsigned)(uintptr_t)&As[0][lr][lseg];
  ldsA[1] = (unsigned)(uintptr_t)&As[1][lr][lseg];
  ldsB[0] = (unsigned)(uintptr_t)&Bs[0][lr][lseg];
  ldsB[1] = (unsigned)(uintptr_t)&Bs[1][lr][lseg];

  // prologue: fill buffer 0
  ASYNC_CP16(ldsA[0],      (const bf16*)(Arow));
  ASYNC_CP16(ldsA[0] + 16, (const bf16*)(Arow + 8));
  ASYNC_CP16(ldsB[0],      (const bf16*)(Brow));
  ASYNC_CP16(ldsB[0] + 16, (const bf16*)(Brow + 8));

  int buf = 0;
  for (int kk = 0; kk < Ksz; kk += 32) {
    if (kk + 32 < Ksz) {
      int nx = buf ^ 1;
      ASYNC_CP16(ldsA[nx],      (const bf16*)(Arow + kk + 32));
      ASYNC_CP16(ldsA[nx] + 16, (const bf16*)(Arow + kk + 40));
      ASYNC_CP16(ldsB[nx],      (const bf16*)(Brow + kk + 32));
      ASYNC_CP16(ldsB[nx] + 16, (const bf16*)(Brow + kk + 40));
      WAIT_ASYNC4();                    // current buffer's 4 copies done
    } else {
      WAIT_ASYNC0();
    }
    __syncthreads();                    // all waves' copies visible

    V16U av[2];
#pragma unroll
    for (int r = 0; r < 2; ++r) {
      int mrow = (wm << 5) + (r << 4) + q16;
      av[r].h[0] = *(const v8bf*)&As[buf][mrow][abase];
      av[r].h[1] = *(const v8bf*)&As[buf][mrow][abase + 16];
    }
    V16U bv[4];
#pragma unroll
    for (int c = 0; c < 4; ++c) {
      int nrow = (wn << 6) + (c << 4) + q16;
      bv[c].h[0] = *(const v8bf*)&Bs[buf][nrow][kboff];
      bv[c].h[1] = *(const v8bf*)&Bs[buf][nrow][kboff + 16];
    }
#pragma unroll
    for (int r = 0; r < 2; ++r)
#pragma unroll
      for (int c = 0; c < 4; ++c)
        acc[r][c] = wmma_bf16(av[r].v, bv[c].v, acc[r][c]);

    __syncthreads();                    // reads done before next overwrite
    buf ^= 1;
  }

#pragma unroll
  for (int r = 0; r < 2; ++r)
#pragma unroll
    for (int c = 0; c < 4; ++c) {
      int gn = n0 + (wn << 6) + (c << 4) + q16;
      float bsv = bias[gn];
#pragma unroll
      for (int e = 0; e < 8; ++e) {
        int gm = m0 + (wm << 5) + (r << 4) + e + (loh ? 0 : 8);
        float val = acc[r][c][e] + bsv;
        if (MODE == 0) {
          int sec = gn >> 10, cc = gn & 1023;
          int h = cc >> 6, dd = cc & 63;
          int bb = gm >> 11, t = gm & 2047;
          size_t bh = (size_t)(bb * Hc + h);
          if (sec == 0)
            qout[(bh * Tc + t) * Dc + dd] = (bf16)(val * 0.125f); // fold softmax scale
          else if (sec == 1)
            kout[(bh * Tc + t) * Dc + dd] = (bf16)val;
          else
            vout[(bh * Dc + dd) * Tc + t] = (bf16)val;            // transposed [d][t]
        } else {
          outf[(size_t)gm * Nsz + gn] = val;
        }
      }
    }
}

// ---------------- flash attention: 1 wave handles 32 query rows ----------------
template<bool MASK>
__device__ __forceinline__ void attn_block(
    int j0, int qrow0, int q16, bool loh, int abase, int kboff,
    const bf16* __restrict__ kbase, const bf16* __restrict__ vt,
    const V16U (&aq)[2][2], v8f (&o)[2][4],
    float (&rmax)[2][8], float (&rsum)[2][8],
    bf16 (*Ps)[32], const v8f& zf)
{
  v8f s[2][2];
#pragma unroll
  for (int nbk = 0; nbk < 2; ++nbk) {
    const bf16* krow = kbase + (size_t)(j0 + (nbk << 4) + q16) * Dc;
    V16U b0, b1;
    b0.h[0] = *(const v8bf*)(krow + kboff);
    b0.h[1] = *(const v8bf*)(krow + kboff + 16);
    b1.h[0] = *(const v8bf*)(krow + 32 + kboff);
    b1.h[1] = *(const v8bf*)(krow + 32 + kboff + 16);
#pragma unroll
    for (int rf = 0; rf < 2; ++rf) {
      s[rf][nbk] = wmma_bf16(aq[rf][0].v, b0.v, zf);
      s[rf][nbk] = wmma_bf16(aq[rf][1].v, b1.v, s[rf][nbk]);
    }
  }
  float sc[2][8];
#pragma unroll
  for (int rf = 0; rf < 2; ++rf) {
#pragma unroll
    for (int e = 0; e < 8; ++e) {
      float x0 = s[rf][0][e], x1 = s[rf][1][e];
      if (MASK) {
        int i = qrow0 + (rf << 4) + e + (loh ? 0 : 8);
        if (j0 + q16 > i)      x0 = -1e30f;
        if (j0 + 16 + q16 > i) x1 = -1e30f;
        s[rf][0][e] = x0; s[rf][1][e] = x1;
      }
      float m = fmaxf(x0, x1);
      m = fmaxf(m, __shfl_xor(m, 1));
      m = fmaxf(m, __shfl_xor(m, 2));
      m = fmaxf(m, __shfl_xor(m, 4));
      m = fmaxf(m, __shfl_xor(m, 8));
      float nm = fmaxf(rmax[rf][e], m);
      sc[rf][e] = __expf(rmax[rf][e] - nm);
      rmax[rf][e] = nm;
      float p0 = __expf(s[rf][0][e] - nm);
      float p1 = __expf(s[rf][1][e] - nm);
      int r = (rf << 4) + e + (loh ? 0 : 8);
      Ps[r][q16]      = (bf16)p0;
      Ps[r][q16 + 16] = (bf16)p1;
      float psv = p0 + p1;
      psv += __shfl_xor(psv, 1);
      psv += __shfl_xor(psv, 2);
      psv += __shfl_xor(psv, 4);
      psv += __shfl_xor(psv, 8);
      rsum[rf][e] = rsum[rf][e] * sc[rf][e] + psv;
    }
  }
  V16U pA[2];
#pragma unroll
  for (int rf = 0; rf < 2; ++rf) {
    pA[rf].h[0] = *(const v8bf*)&Ps[(rf << 4) + q16][abase];
    pA[rf].h[1] = *(const v8bf*)&Ps[(rf << 4) + q16][abase + 16];
  }
#pragma unroll
  for (int n4 = 0; n4 < 4; ++n4) {
    const bf16* vrow = vt + (size_t)((n4 << 4) + q16) * Tc + j0 + kboff;
    V16U bV;
    bV.h[0] = *(const v8bf*)(vrow);
    bV.h[1] = *(const v8bf*)(vrow + 16);
#pragma unroll
    for (int rf = 0; rf < 2; ++rf) {
#pragma unroll
      for (int e = 0; e < 8; ++e) o[rf][n4][e] *= sc[rf][e];
      o[rf][n4] = wmma_bf16(pA[rf].v, bV.v, o[rf][n4]);
    }
  }
}

__global__ __launch_bounds__(32)
void k_flash_attn(const bf16* __restrict__ qb, const bf16* __restrict__ kb,
                  const bf16* __restrict__ vtb, bf16* __restrict__ yb)
{
  __shared__ bf16 Ps[32][32];
  int lane = threadIdx.x;
  bool loh = lane < 16;
  int q16 = lane & 15;
  int abase = loh ? 0 : 8;
  int kboff = loh ? 0 : 8;
  int bh = blockIdx.x >> 6;          // 0..31  (b*H+h)
  int rb = blockIdx.x & 63;          // 64 row-blocks of 32 rows
  int qrow0 = rb << 5;
  const size_t base = (size_t)bh * Tc * Dc;
  const bf16* kbase = kb + base;
  const bf16* vt = vtb + base;       // [d][t]

  // Q A-fragments: 2 row-frags x (2 k-chunks of 32 dims); q pre-scaled by 0.125
  V16U aq[2][2];
#pragma unroll
  for (int rf = 0; rf < 2; ++rf) {
    const bf16* qrow = qb + base + (size_t)(qrow0 + (rf << 4) + q16) * Dc;
    aq[rf][0].h[0] = *(const v8bf*)(qrow + abase);
    aq[rf][0].h[1] = *(const v8bf*)(qrow + abase + 16);
    aq[rf][1].h[0] = *(const v8bf*)(qrow + 32 + abase);
    aq[rf][1].h[1] = *(const v8bf*)(qrow + 32 + abase + 16);
  }

  v8f o[2][4];
  float rmax[2][8], rsum[2][8];
#pragma unroll
  for (int rf = 0; rf < 2; ++rf) {
#pragma unroll
    for (int n4 = 0; n4 < 4; ++n4)
#pragma unroll
      for (int e = 0; e < 8; ++e) o[rf][n4][e] = 0.f;
#pragma unroll
    for (int e = 0; e < 8; ++e) { rmax[rf][e] = -1e30f; rsum[rf][e] = 0.f; }
  }
  v8f zf;
#pragma unroll
  for (int e = 0; e < 8; ++e) zf[e] = 0.f;

  int j0 = 0;
  for (; j0 + 31 <= qrow0; j0 += 32)   // fully-unmasked key blocks
    attn_block<false>(j0, qrow0, q16, loh, abase, kboff, kbase, vt, aq, o, rmax, rsum, Ps, zf);
  for (; j0 <= qrow0 + 31; j0 += 32)   // single diagonal (masked) block
    attn_block<true>(j0, qrow0, q16, loh, abase, kboff, kbase, vt, aq, o, rmax, rsum, Ps, zf);

  // ---- epilogue: normalize and store y (bf16, head-major) ----
#pragma unroll
  for (int rf = 0; rf < 2; ++rf) {
#pragma unroll
    for (int e = 0; e < 8; ++e) rsum[rf][e] = 1.f / rsum[rf][e];
#pragma unroll
    for (int n4 = 0; n4 < 4; ++n4)
#pragma unroll
      for (int e = 0; e < 8; ++e) {
        int i = qrow0 + (rf << 4) + e + (loh ? 0 : 8);
        yb[base + (size_t)i * Dc + (n4 << 4) + q16] = (bf16)(o[rf][n4][e] * rsum[rf][e]);
      }
  }
}

// ---------------- KNN memory attention + gated combine ----------------
__global__ __launch_bounds__(128)
void k_memattn(const bf16* __restrict__ qb, const bf16* __restrict__ yb,
               const float* __restrict__ db, const int* __restrict__ indices,
               const float* __restrict__ gate, bf16* __restrict__ comb)
{
  int token = blockIdx.x;            // b*T + t
  int b = token >> 11, t = token & 2047;
  int tid = threadIdx.x;
  int h = tid >> 3, sub = tid & 7;   // 8 lanes per head, 8 dims each
  size_t hoff = (((size_t)(b * Hc + h)) * Tc + t) * Dc + sub * 8;

  float qv[8];
  const bf16* qp = qb + hoff;        // q pre-scaled by 0.125 (== mem scale)
#pragma unroll
  for (int e = 0; e < 8; ++e) qv[e] = (float)qp[e];

  int idx0 = indices[token * 3 + 0];
  int idx1 = indices[token * 3 + 1];
  int idx2 = indices[token * 3 + 2];
  int idxs[3] = {idx0, idx1, idx2};

  float score[3];
#pragma unroll
  for (int k = 0; k < 3; ++k) {
    const float* kr = db + ((size_t)idxs[k] * 2) * Cc + h * Dc + sub * 8;
    float p = 0.f;
#pragma unroll
    for (int e = 0; e < 8; ++e) p += qv[e] * kr[e];
    p += __shfl_xor(p, 1);
    p += __shfl_xor(p, 2);
    p += __shfl_xor(p, 4);
    score[k] = p;
  }
  float mx = fmaxf(score[0], fmaxf(score[1], score[2]));
  float e0 = __expf(score[0] - mx);
  float e1 = __expf(score[1] - mx);
  float e2 = __expf(score[2] - mx);
  float inv = 1.f / (e0 + e1 + e2);
  float att[3] = {e0 * inv, e1 * inv, e2 * inv};

  float acc[8];
#pragma unroll
  for (int e = 0; e < 8; ++e) acc[e] = 0.f;
#pragma unroll
  for (int k = 0; k < 3; ++k) {
    const float* vr = db + ((size_t)idxs[k] * 2 + 1) * Cc + h * Dc + sub * 8;
#pragma unroll
    for (int e = 0; e < 8; ++e) acc[e] += att[k] * vr[e];
  }
  float g = gate[h];
  const bf16* yp = yb + hoff;
  bf16* cp = comb + (size_t)token * Cc + h * Dc + sub * 8;
#pragma unroll
  for (int e = 0; e < 8; ++e)
    cp[e] = (bf16)(acc[e] * g + (float)yp[e] * (1.f - g));
}

// ---------------- host launch ----------------
extern "C" void kernel_launch(void* const* d_in, const int* in_sizes, int n_in,
                              void* d_out, int out_size, void* d_ws, size_t ws_size,
                              hipStream_t stream)
{
  (void)in_sizes; (void)n_in; (void)out_size; (void)ws_size;
  const float* x     = (const float*)d_in[0];
  const float* Wqkv  = (const float*)d_in[1];
  const float* bqkv  = (const float*)d_in[2];
  const float* Wproj = (const float*)d_in[3];
  const float* bproj = (const float*)d_in[4];
  const float* gate  = (const float*)d_in[5];
  const float* db    = (const float*)d_in[6];
  const int*   idx   = (const int*)d_in[7];
  float* out = (float*)d_out;

  char* ws = (char*)d_ws;
  size_t off = 0;
  auto take = [&](size_t bytes) -> char* {
    char* p = ws + off;
    off = (off + bytes + 255) & ~(size_t)255;
    return p;
  };
  bf16* xb     = (bf16*)take((size_t)Mrows * Cc * 2);          // 8 MB
  bf16* wqkvT  = (bf16*)take((size_t)3 * Cc * Cc * 2);         // 6 MB
  bf16* wprojT = (bf16*)take((size_t)Cc * Cc * 2);             // 2 MB
  bf16* qbuf   = (bf16*)take((size_t)Bc * Hc * Tc * Dc * 2);   // 8 MB
  bf16* kbuf   = (bf16*)take((size_t)Bc * Hc * Tc * Dc * 2);   // 8 MB
  bf16* vtbuf  = (bf16*)take((size_t)Bc * Hc * Tc * Dc * 2);   // 8 MB (transposed)
  bf16* ybuf   = (bf16*)take((size_t)Bc * Hc * Tc * Dc * 2);   // 8 MB
  bf16* comb   = (bf16*)take((size_t)Mrows * Cc * 2);          // 8 MB

  int nx = Mrows * Cc;
  k_f32_to_bf16<<<(nx + 255) / 256, 256, 0, stream>>>(x, xb, nx);
  k_transpose_bf16<<<dim3(3 * Cc / 32, Cc / 32), dim3(32, 8), 0, stream>>>(Wqkv, wqkvT, Cc, 3 * Cc);
  k_transpose_bf16<<<dim3(Cc / 32, Cc / 32), dim3(32, 8), 0, stream>>>(Wproj, wprojT, Cc, Cc);

  k_wmma_gemm<0><<<(Mrows / 128) * (3 * Cc / 128), 256, 0, stream>>>(
      xb, wqkvT, bqkv, 3 * Cc, Cc, nullptr, qbuf, kbuf, vtbuf);

  k_flash_attn<<<Bc * Hc * (Tc / 32), 32, 0, stream>>>(qbuf, kbuf, vtbuf, ybuf);

  k_memattn<<<Bc * Tc, 128, 0, stream>>>(qbuf, ybuf, db, idx, gate, comb);

  k_wmma_gemm<1><<<(Mrows / 128) * (Cc / 128), 256, 0, stream>>>(
      comb, wprojT, bproj, Cc, Cc, out, nullptr, nullptr, nullptr);
}
